// GlobalAggregationBlock_1039382086104
// MI455X (gfx1250) — matrix operand order repaired and checked by reading, not compiled, not run on hardware
//
#include <hip/hip_runtime.h>
#include <math.h>

// ---------------------------------------------------------------------------
// GlobalAggregationBlock for MI455X (gfx1250, wave32, WMMA + async LDS copy).
//   x:(2,64,20,20,20) -> qkv proj -> flash attention (N=8000, d=64) -> wo proj
// Attention matmuls: v_wmma_f32_16x16x32_f16 (f32 accumulate).
// K/V tiles double-buffered in LDS via GLOBAL_LOAD_ASYNC_TO_LDS_B128 and
// shared by 4 waves x 32 queries = 128 queries per workgroup.
// ---------------------------------------------------------------------------

typedef __attribute__((ext_vector_type(16))) _Float16 v16h;
typedef __attribute__((ext_vector_type(8)))  _Float16 v8h;
typedef __attribute__((ext_vector_type(8)))  float    v8f;
typedef int v4i __attribute__((vector_size(16)));

#define AS1 __attribute__((address_space(1)))
#define AS3 __attribute__((address_space(3)))

#define NVOX 8000            // D*H*W
#define NB   2               // batch
#define CCH  64              // channels / head dim
#define NPAD 8192            // query dim padded so 128-query WGs divide evenly
#define QPB  128             // queries per workgroup (4 waves x 32)
#define NITER (NVOX / 32)    // 250 key-block iterations

// LDS tile layout (halves), rows padded to spread bank accesses
#define KROW 72              // 64 payload + 8 pad
#define VROW 40              // 32 payload + 8 pad
#define KTILE_H (32 * KROW)  // 2304 halves
#define VTILE_H (64 * VROW)  // 2560 halves
#define BUFH (KTILE_H + VTILE_H)

#if __has_builtin(__builtin_amdgcn_global_load_async_to_lds_b128)
#define HAS_ASYNC_LDS 1
#else
#define HAS_ASYNC_LDS 0
#endif

static __device__ inline v8h ld8h(const _Float16* p) { return *(const v8h*)p; }

static __device__ inline v16h cat16(v8h lo, v8h hi) {
  v16h r;
#pragma unroll
  for (int i = 0; i < 8; ++i) { r[i] = lo[i]; r[8 + i] = hi[i]; }
  return r;
}

static __device__ inline v8f wmma_f16(v16h a, v16h b, v8f c) {
  return __builtin_amdgcn_wmma_f32_16x16x32_f16(
      false, a, false, b, (short)0, c, false, false);
}

// 16-byte global -> LDS copy (async on CDNA5; sync fallback otherwise)
static __device__ inline void cp16(const _Float16* g, _Float16* l) {
#if HAS_ASYNC_LDS
  AS1 v4i* gp = (AS1 v4i*)(AS1 void*)(_Float16*)g;
  AS3 v4i* lp = (AS3 v4i*)(AS3 void*)l;
  __builtin_amdgcn_global_load_async_to_lds_b128(gp, lp, 0, 0);
#else
  *(uint4*)l = *(const uint4*)g;
#endif
}

// Stage one 32-key K tile (32x64) + V tile (64x32) into LDS. 128 threads,
// 4 x 16B chunks per thread -> 4 async-load instructions per wave.
static __device__ inline void stage_tiles(const _Float16* __restrict__ Kb,
                                          const _Float16* __restrict__ Vb,
                                          _Float16* __restrict__ kt,
                                          _Float16* __restrict__ vt,
                                          int kb, int tid)
{
#pragma unroll
  for (int i = 0; i < 2; ++i) {              // K: 32 rows x 8 chunks = 256
    const int c = tid + i * 128;
    const int kr = c >> 3, kc = c & 7;
    cp16(Kb + (size_t)(kb + kr) * CCH + kc * 8, kt + kr * KROW + kc * 8);
  }
#pragma unroll
  for (int i = 0; i < 2; ++i) {              // V: 64 dims x 4 chunks = 256
    const int c = tid + i * 128;
    const int vd = c >> 2, vp = c & 3;
    cp16(Vb + (size_t)vd * NVOX + kb + vp * 8, vt + vd * VROW + vp * 8);
  }
}

// ---------------------------------------------------------------------------
// Kernel 1: fused QKV projection. One thread per (padded) voxel.
//   Qh : (B, NPAD, 64) f16, pre-scaled by 1/sqrt(64); pad rows zeroed
//   Kh : (B, N, 64) f16
//   Vt : (B, 64, N) f16
// ---------------------------------------------------------------------------
__global__ __launch_bounds__(256)
void qkv_proj_kernel(const float* __restrict__ x,
                     const float* __restrict__ wq, const float* __restrict__ bq,
                     const float* __restrict__ wk, const float* __restrict__ bk,
                     const float* __restrict__ wv, const float* __restrict__ bv,
                     _Float16* __restrict__ Qh, _Float16* __restrict__ Kh,
                     _Float16* __restrict__ Vt)
{
  const int t = blockIdx.x * blockDim.x + threadIdx.x;
  if (t >= NB * NPAD) return;
  const int b = t / NPAD;
  const int n = t % NPAD;

  _Float16* qrow = Qh + ((size_t)b * NPAD + n) * CCH;
  if (n >= NVOX) {                    // zero the query padding rows
#pragma unroll
    for (int o = 0; o < CCH; ++o) qrow[o] = (_Float16)0.0f;
    return;
  }

  const float* xb = x + (size_t)b * CCH * NVOX + n;
  float xv[CCH];
#pragma unroll
  for (int c = 0; c < CCH; ++c) xv[c] = xb[(size_t)c * NVOX];

  _Float16* krow = Kh + ((size_t)b * NVOX + n) * CCH;
  _Float16* vcol = Vt + (size_t)b * CCH * NVOX + n;

  for (int o = 0; o < CCH; ++o) {
    float aq = bq[o], ak = bk[o], av = bv[o];
    const float* wqr = wq + o * CCH;
    const float* wkr = wk + o * CCH;
    const float* wvr = wv + o * CCH;
#pragma unroll
    for (int c = 0; c < CCH; ++c) {
      const float xc = xv[c];
      aq = fmaf(wqr[c], xc, aq);
      ak = fmaf(wkr[c], xc, ak);
      av = fmaf(wvr[c], xc, av);
    }
    qrow[o] = (_Float16)(aq * 0.125f);   // fold 1/sqrt(64) into Q
    krow[o] = (_Float16)ak;
    vcol[(size_t)o * NVOX] = (_Float16)av;
  }
}

// ---------------------------------------------------------------------------
// Kernel 2: flash attention. 4 waves/WG, 32 queries/wave, K/V tiles shared
// through double-buffered LDS. S^T = K x Q^T so exp'd scores are already in
// the A-operand layout of the O += P x V WMMA (no cross-lane movement).
// ---------------------------------------------------------------------------
__global__ __launch_bounds__(128)
void attn_kernel(const _Float16* __restrict__ Qh,
                 const _Float16* __restrict__ Kh,
                 const _Float16* __restrict__ Vt,
                 float* __restrict__ Obuf)
{
  __shared__ _Float16 smem[2 * BUFH];     // 19456 bytes

  const int tid  = threadIdx.x;
  const int wave = tid >> 5;
  const int lane = tid & 31;
  const int q16  = lane & 15;
  const bool hi  = lane >= 16;
  const int b     = blockIdx.x / (NPAD / QPB);
  const int blk   = blockIdx.x % (NPAD / QPB);
  const int qbase = blk * QPB + wave * 32;   // this wave's 32 queries

  const _Float16* Qb = Qh + (size_t)b * NPAD * CCH;
  const _Float16* Kb = Kh + (size_t)b * NVOX * CCH;
  const _Float16* Vb = Vt + (size_t)b * CCH * NVOX;

  // Q as B-operand (32x16): lane q holds channels 8h..8h+7 and 16+8h..16+8h+7 of query q.
  v16h qb[2][2];
#pragma unroll
  for (int t = 0; t < 2; ++t) {
    const _Float16* qrow = Qb + (size_t)(qbase + t * 16 + q16) * CCH + (hi ? 8 : 0);
    qb[t][0] = cat16(ld8h(qrow), ld8h(qrow + 16));
    qb[t][1] = cat16(ld8h(qrow + 32), ld8h(qrow + 48));
  }

  v8f o[2][4] = {};
  float mrun[2] = { -__builtin_inff(), -__builtin_inff() };
  float lrun[2] = { 0.0f, 0.0f };

  stage_tiles(Kb, Vb, smem, smem + KTILE_H, 0, tid);

  for (int it = 0; it < NITER; ++it) {
    const int cur = it & 1;
    const _Float16* kt = smem + cur * BUFH;
    const _Float16* vt = kt + KTILE_H;

    if (it + 1 < NITER) {
      _Float16* kn = smem + (1 - cur) * BUFH;
      stage_tiles(Kb, Vb, kn, kn + KTILE_H, (it + 1) * 32, tid);
#if HAS_ASYNC_LDS
      asm volatile("s_wait_asynccnt 0x4" ::: "memory");  // cur's 4 copies done
#endif
    } else {
#if HAS_ASYNC_LDS
      asm volatile("s_wait_asynccnt 0x0" ::: "memory");
#endif
    }
    __syncthreads();   // cur tile visible to all waves

    // K A-operands (16 keys x 32 ch each), shared by both query tiles
    const _Float16* k0 = kt + q16 * KROW + (hi ? 8 : 0);
    const _Float16* k1 = kt + (16 + q16) * KROW + (hi ? 8 : 0);
    const v16h a00 = cat16(ld8h(k0),      ld8h(k0 + 16));   // keys 0-15, ch 0-31
    const v16h a01 = cat16(ld8h(k0 + 32), ld8h(k0 + 48));   // keys 0-15, ch 32-63
    const v16h a10 = cat16(ld8h(k1),      ld8h(k1 + 16));   // keys 16-31, ch 0-31
    const v16h a11 = cat16(ld8h(k1 + 32), ld8h(k1 + 48));   // keys 16-31, ch 32-63

    // V B-operands (32 keys x 16 dims), shared by both query tiles
    v16h vbo[4];
#pragma unroll
    for (int c = 0; c < 4; ++c) {
      const _Float16* vp = vt + (c * 16 + q16) * VROW + (hi ? 8 : 0);
      vbo[c] = cat16(ld8h(vp), ld8h(vp + 16));
    }

#pragma unroll
    for (int t = 0; t < 2; ++t) {
      // scores S^T (keys x queries), f32 accumulate over 64 channels
      v8f z = {};
      v8f st0 = wmma_f16(a00, qb[t][0], z);
      st0     = wmma_f16(a01, qb[t][1], st0);
      v8f st1 = wmma_f16(a10, qb[t][0], z);
      st1     = wmma_f16(a11, qb[t][1], st1);

      // online softmax: 16 keys in-lane, fold halves with shfl_xor(16)
      float lm = st0[0];
#pragma unroll
      for (int r = 1; r < 8; ++r) lm = fmaxf(lm, st0[r]);
#pragma unroll
      for (int r = 0; r < 8; ++r) lm = fmaxf(lm, st1[r]);
      lm = fmaxf(lm, __shfl_xor(lm, 16));
      const float mn   = fmaxf(mrun[t], lm);
      const float corr = __expf(mrun[t] - mn);

      v8f p0, p1;
      float ls = 0.0f;
#pragma unroll
      for (int r = 0; r < 8; ++r) {
        p0[r] = __expf(st0[r] - mn);
        p1[r] = __expf(st1[r] - mn);
        ls += p0[r] + p1[r];
      }
      ls += __shfl_xor(ls, 16);
      lrun[t] = lrun[t] * corr + ls;
      mrun[t] = mn;

      // rescale O accumulators (broadcast corr of the row each comp holds)
#pragma unroll
      for (int r = 0; r < 8; ++r) {
        const float cr = __shfl(corr, hi ? (r + 8) : r);
        o[t][0][r] *= cr; o[t][1][r] *= cr;
        o[t][2][r] *= cr; o[t][3][r] *= cr;
      }

      // P (queries x 32 keys) already in A layout: convert to f16
      v16h ap;
#pragma unroll
      for (int r = 0; r < 8; ++r) {
        ap[r]     = (_Float16)p0[r];
        ap[8 + r] = (_Float16)p1[r];
      }

#pragma unroll
      for (int c = 0; c < 4; ++c)
        o[t][c] = wmma_f16(ap, vbo[c], o[t][c]);
    }
    __syncthreads();   // all waves done with cur before it is overwritten
  }

  // normalize and store O: (B, N, 64) f32, flat (B, 512000)
  float* ob = Obuf + (size_t)b * NVOX * CCH;
#pragma unroll
  for (int t = 0; t < 2; ++t) {
    const float linv = 1.0f / lrun[t];
#pragma unroll
    for (int r = 0; r < 8; ++r) {
      const float lr = __shfl(linv, hi ? (r + 8) : r);
      const int row = qbase + t * 16 + r + (hi ? 8 : 0);
      if (row < NVOX) {
        float* rp = ob + (size_t)row * CCH + q16;
        rp[0]  = o[t][0][r] * lr;
        rp[16] = o[t][1][r] * lr;
        rp[32] = o[t][2][r] * lr;
        rp[48] = o[t][3][r] * lr;
      }
    }
  }
}

// ---------------------------------------------------------------------------
// Kernel 3: output projection. The reference reinterprets (B,N,64) memory as
// (B,64,N) before applying wo, so we read Obuf flat at stride N.
// ---------------------------------------------------------------------------
__global__ __launch_bounds__(256)
void out_proj_kernel(const float* __restrict__ Obuf,
                     const float* __restrict__ wo, const float* __restrict__ bo,
                     float* __restrict__ out)
{
  const int t = blockIdx.x * blockDim.x + threadIdx.x;
  if (t >= NB * NVOX) return;
  const int b = t / NVOX;
  const int j = t % NVOX;

  const float* ib = Obuf + (size_t)b * CCH * NVOX + j;
  float ov[CCH];
#pragma unroll
  for (int c = 0; c < CCH; ++c) ov[c] = ib[(size_t)c * NVOX];

  float* op = out + (size_t)b * CCH * NVOX + j;
  for (int o = 0; o < CCH; ++o) {
    float acc = bo[o];
    const float* wr = wo + o * CCH;
#pragma unroll
    for (int c = 0; c < CCH; ++c) acc = fmaf(wr[c], ov[c], acc);
    op[(size_t)o * NVOX] = acc;
  }
}

// ---------------------------------------------------------------------------
extern "C" void kernel_launch(void* const* d_in, const int* in_sizes, int n_in,
                              void* d_out, int out_size, void* d_ws, size_t ws_size,
                              hipStream_t stream) {
  const float* x  = (const float*)d_in[0];
  const float* wq = (const float*)d_in[1];
  const float* bq = (const float*)d_in[2];
  const float* wk = (const float*)d_in[3];
  const float* bk = (const float*)d_in[4];
  const float* wv = (const float*)d_in[5];
  const float* bv = (const float*)d_in[6];
  const float* wo = (const float*)d_in[7];
  const float* bo = (const float*)d_in[8];
  float* out = (float*)d_out;

  // workspace layout (bytes):
  //   Qh (B,NPAD,64) f16 : 2,097,152
  //   Kh (B,N,64)    f16 : 2,048,000
  //   Vt (B,64,N)    f16 : 2,048,000
  //   Obuf (B,N,64)  f32 : 4,096,000     total ~10.3 MB
  char* ws = (char*)d_ws;
  const size_t qBytes = (size_t)NB * NPAD * CCH * sizeof(_Float16);
  const size_t kBytes = (size_t)NB * NVOX * CCH * sizeof(_Float16);
  _Float16* Qh  = (_Float16*)(ws);
  _Float16* Kh  = (_Float16*)(ws + qBytes);
  _Float16* Vt  = (_Float16*)(ws + qBytes + kBytes);
  float*    Obf = (float*)   (ws + qBytes + 2 * kBytes);

  qkv_proj_kernel<<<(NB * NPAD + 255) / 256, 256, 0, stream>>>(
      x, wq, bq, wk, bk, wv, bv, Qh, Kh, Vt);

  attn_kernel<<<NB * (NPAD / QPB), 128, 0, stream>>>(Qh, Kh, Vt, Obf);

  out_proj_kernel<<<(NB * NVOX + 255) / 256, 256, 0, stream>>>(Obf, wo, bo, out);
}
